// GIN_GRU_88802743812677
// MI455X (gfx1250) — matrix operand order repaired and checked by reading, not compiled, not run on hardware
//
#include <hip/hip_runtime.h>
#include <hip/hip_bf16.h>

typedef _Float16 v16h __attribute__((ext_vector_type(16)));
typedef float    v8f  __attribute__((ext_vector_type(8)));

#define N_NODES  50000
#define ROWTILES 3125      // 50000 / 16
#define NEDGE    800000

__device__ __forceinline__ v8f wmma_f16(v16h a, v16h b, v8f c) {
  // D = A(16x32 f16) * B(32x16 f16) + C(16x16 f32)
  return __builtin_amdgcn_wmma_f32_16x16x32_f16(false, a, false, b, (short)0, c,
                                                false, false);
}

// Load one lane's A-fragment (16 f16 values) from an f32 LDS tile.
// A layout: lanes 0-15 M=lane, K = {kb..kb+7, kb+16..kb+23}; lanes 16-31 kb += 8.
__device__ __forceinline__ v16h load_a_f32(const float* ap) {
  float4 q0 = *(const float4*)(ap);
  float4 q1 = *(const float4*)(ap + 4);
  float4 q2 = *(const float4*)(ap + 16);
  float4 q3 = *(const float4*)(ap + 20);
  v16h a;
  a[0]=(_Float16)q0.x;  a[1]=(_Float16)q0.y;  a[2]=(_Float16)q0.z;  a[3]=(_Float16)q0.w;
  a[4]=(_Float16)q1.x;  a[5]=(_Float16)q1.y;  a[6]=(_Float16)q1.z;  a[7]=(_Float16)q1.w;
  a[8]=(_Float16)q2.x;  a[9]=(_Float16)q2.y;  a[10]=(_Float16)q2.z; a[11]=(_Float16)q2.w;
  a[12]=(_Float16)q3.x; a[13]=(_Float16)q3.y; a[14]=(_Float16)q3.z; a[15]=(_Float16)q3.w;
  return a;
}

// One wave computes one 16x16 f32 tile: acc = LDS_A(16 x NK*32) @ Wp-tile.
// wp must already point at this wave's column tile (ntile*NK*512 halves).
template <int NK>
__device__ __forceinline__ v8f wave_gemm(const float* sBase, int stride,
                                         const _Float16* wp, int lane) {
  const int lo = lane & 15, hi = lane >> 4;
  v8f acc = {};
  #pragma unroll
  for (int kt = 0; kt < NK; ++kt) {
    v16h a = load_a_f32(sBase + lo * stride + kt * 32 + hi * 8);
    v16h b = *(const v16h*)(wp + (size_t)kt * 512 + lane * 16);
    acc = wmma_f16(a, b, acc);
  }
  return acc;
}

// ---------------------------------------------------------------------------
// Pack a f32 weight matrix (Din x Dout; or transposed source Dout x Din) into
// the WMMA B-operand lane layout, converted to f16.
// dst index = ((ntile*nK + ktile)*32 + lane)*16 + e
//   lanes 0-15 : N = ntile*16+lane,    K = ktile*32 + {0..7, 16..23}[e]
//   lanes16-31 : N = ntile*16+lane-16, K = ktile*32 + {8..15, 24..31}[e]
// ---------------------------------------------------------------------------
__global__ void pack_w(const float* __restrict__ src, _Float16* __restrict__ dst,
                       int nK, int nT, int transposed) {
  int idx = blockIdx.x * blockDim.x + threadIdx.x;
  int total = nK * nT * 512;
  if (idx >= total) return;
  int e    = idx & 15;
  int lane = (idx >> 4) & 31;
  int rest = idx >> 9;
  int kt   = rest % nK;
  int nt   = rest / nK;
  int n = nt * 16 + (lane & 15);
  int k = kt * 32 + ((lane >> 4) << 3) + (e & 7) + ((e >> 3) << 4);
  int Din = nK * 32, Dout = nT * 16;
  float v = transposed ? src[(size_t)n * Din + k] : src[(size_t)k * Dout + n];
  dst[idx] = (_Float16)v;
}

__global__ void zero_f(float* __restrict__ p, int n) {
  int i = blockIdx.x * blockDim.x + threadIdx.x;
  if (i < n) p[i] = 0.0f;
}

// ---------------------------------------------------------------------------
// mlp0 fused: out = tanh(tanh(tanh(x@W0+b0)@W1+b1)@W2+b2), all 128-wide.
// One global read of x, one global write of out per 16-row tile.
// ---------------------------------------------------------------------------
__global__ void __launch_bounds__(256)
mlp3_fused(const float* __restrict__ A,
           const _Float16* __restrict__ W0, const float* __restrict__ b0,
           const _Float16* __restrict__ W1, const float* __restrict__ b1,
           const _Float16* __restrict__ W2, const float* __restrict__ b2,
           float* __restrict__ out) {
  __shared__ float sA[16 * 132];
  __shared__ float sB[16 * 132];
  __shared__ float sC[16 * 132];
  const int row0 = blockIdx.x * 16;

  for (int j = threadIdx.x; j < 512; j += 256) {       // float4 granules
    int row = j >> 5, k = (j & 31) << 2;
    *(float4*)&sA[row * 132 + k] = *(const float4*)(A + (size_t)(row0 + row) * 128 + k);
  }
  __syncthreads();

  const int wave = threadIdx.x >> 5, lane = threadIdx.x & 31;
  const int lo = lane & 15, hi = lane >> 4;
  const int col = wave * 16 + lo;

  v8f acc = wave_gemm<4>(sA, 132, W0 + (size_t)wave * 2048, lane);
  float bv = b0[col];
  #pragma unroll
  for (int r = 0; r < 8; ++r) sB[(r + hi * 8) * 132 + col] = tanhf(acc[r] + bv);
  __syncthreads();

  acc = wave_gemm<4>(sB, 132, W1 + (size_t)wave * 2048, lane);
  bv = b1[col];
  #pragma unroll
  for (int r = 0; r < 8; ++r) sC[(r + hi * 8) * 132 + col] = tanhf(acc[r] + bv);
  __syncthreads();

  acc = wave_gemm<4>(sC, 132, W2 + (size_t)wave * 2048, lane);
  bv = b2[col];
  #pragma unroll
  for (int r = 0; r < 8; ++r)
    out[(size_t)(row0 + r + hi * 8) * 128 + col] = tanhf(acc[r] + bv);
}

// ---------------------------------------------------------------------------
// One wave per edge: agg[row] += val * t0[col]  (128 floats, 4 per lane).
// ---------------------------------------------------------------------------
__global__ void __launch_bounds__(256)
scatter_edges(const float* __restrict__ t0, const int* __restrict__ erow,
              const int* __restrict__ ecol, const float* __restrict__ eval_,
              float* __restrict__ agg, int E) {
  int wid  = (blockIdx.x * 256 + threadIdx.x) >> 5;
  int lane = threadIdx.x & 31;
  if (wid >= E) return;
  int   dst = erow[wid];
  int   src = ecol[wid];
  float w   = eval_[wid];
  float4 v = *((const float4*)(t0 + (size_t)src * 128) + lane);
  float* d = agg + (size_t)dst * 128 + lane * 4;
  atomicAdd(d + 0, w * v.x);
  atomicAdd(d + 1, w * v.y);
  atomicAdd(d + 2, w * v.z);
  atomicAdd(d + 3, w * v.w);
}

// ---------------------------------------------------------------------------
// Mega-fused per 16-row tile:
//   t1    = mlp1(x)                       (3 tanh layers, LDS-resident)
//   x_new = mlp2(concat(t1, agg))         (3 tanh layers, LDS-resident)
//   h     = GRU(x_new, h)                 (48 WMMA tiles -> gates in LDS)
//   xout  = h / std(h, ddof=1)
// x_new never touches global memory.
// ---------------------------------------------------------------------------
__global__ void __launch_bounds__(256)
mega_fused(const float* __restrict__ x, const float* __restrict__ agg,
           float* __restrict__ h,
           const _Float16* __restrict__ W10, const float* __restrict__ B10,
           const _Float16* __restrict__ W11, const float* __restrict__ B11,
           const _Float16* __restrict__ W12, const float* __restrict__ B12,
           const _Float16* __restrict__ W20, const float* __restrict__ B20,
           const _Float16* __restrict__ W21, const float* __restrict__ B21,
           const _Float16* __restrict__ W22, const float* __restrict__ B22,
           const _Float16* __restrict__ Wih, const _Float16* __restrict__ Whh,
           const float* __restrict__ bih, const float* __restrict__ bhh,
           float* __restrict__ xout) {
  __shared__ float sA[16 * 260];   // [x | agg] -> later [t1 | agg] (concat input)
  __shared__ float sB[16 * 132];   // ping buffer; finally holds x_new
  __shared__ float hs[16 * 132];   // staged h
  __shared__ float giS[16 * 388];  // GEMM scratch, then gi, then h_new
  __shared__ float ghS[16 * 388];  // gh
  const int row0 = blockIdx.x * 16;

  for (int j = threadIdx.x; j < 512; j += 256) {       // float4 granules
    int row = j >> 5, k = (j & 31) << 2;
    *(float4*)&sA[row * 260 + k]       = *(const float4*)(x   + (size_t)(row0 + row) * 128 + k);
    *(float4*)&sA[row * 260 + 128 + k] = *(const float4*)(agg + (size_t)(row0 + row) * 128 + k);
    *(float4*)&hs[row * 132 + k]       = *(const float4*)(h   + (size_t)(row0 + row) * 128 + k);
  }
  __syncthreads();

  const int wave = threadIdx.x >> 5, lane = threadIdx.x & 31;
  const int lo = lane & 15, hi = lane >> 4;
  const int col = wave * 16 + lo;

  // ---- mlp1: x -> t1 (t1 lands in sA cols 0..127, forming the concat) ----
  v8f acc = wave_gemm<4>(sA, 260, W10 + (size_t)wave * 2048, lane);
  float bv = B10[col];
  #pragma unroll
  for (int r = 0; r < 8; ++r) sB[(r + hi * 8) * 132 + col] = tanhf(acc[r] + bv);
  __syncthreads();

  acc = wave_gemm<4>(sB, 132, W11 + (size_t)wave * 2048, lane);
  bv = B11[col];
  #pragma unroll
  for (int r = 0; r < 8; ++r) giS[(r + hi * 8) * 388 + col] = tanhf(acc[r] + bv);
  __syncthreads();

  acc = wave_gemm<4>(giS, 388, W12 + (size_t)wave * 2048, lane);
  bv = B12[col];
  #pragma unroll
  for (int r = 0; r < 8; ++r) sA[(r + hi * 8) * 260 + col] = tanhf(acc[r] + bv);
  __syncthreads();

  // ---- mlp2: concat(t1, agg) [256-wide] -> x_new (ends in sB) ----
  acc = wave_gemm<8>(sA, 260, W20 + (size_t)wave * 4096, lane);
  bv = B20[col];
  #pragma unroll
  for (int r = 0; r < 8; ++r) sB[(r + hi * 8) * 132 + col] = tanhf(acc[r] + bv);
  __syncthreads();

  acc = wave_gemm<4>(sB, 132, W21 + (size_t)wave * 2048, lane);
  bv = B21[col];
  #pragma unroll
  for (int r = 0; r < 8; ++r) giS[(r + hi * 8) * 388 + col] = tanhf(acc[r] + bv);
  __syncthreads();

  acc = wave_gemm<4>(giS, 388, W22 + (size_t)wave * 2048, lane);
  bv = B22[col];
  #pragma unroll
  for (int r = 0; r < 8; ++r) sB[(r + hi * 8) * 132 + col] = tanhf(acc[r] + bv);
  __syncthreads();

  // ---- GRU: gi = x_new @ Wih^T + bih, gh = h @ Whh^T + bhh (48 tiles / 8 waves)
  for (int t = wave; t < 48; t += 8) {
    const bool isH = (t >= 24);
    const int  ct  = isH ? (t - 24) : t;
    const float*    As = isH ? hs : sB;
    const _Float16* wp = (isH ? Whh : Wih) + (size_t)ct * 2048;
    v8f a2 = wave_gemm<4>(As, 132, wp, lane);
    const float bv2 = (isH ? bhh : bih)[ct * 16 + lo];
    float* dst = isH ? ghS : giS;
    #pragma unroll
    for (int r = 0; r < 8; ++r)
      dst[(r + hi * 8) * 388 + ct * 16 + lo] = a2[r] + bv2;
  }
  __syncthreads();

  // ---- gates (order r, z, n); each c<128 slot is read+written by its owner only
  #pragma unroll
  for (int i = 0; i < 8; ++i) {
    int id  = threadIdx.x + i * 256;
    int row = id >> 7, c = id & 127;
    float ir  = giS[row * 388 + c];
    float iz  = giS[row * 388 + 128 + c];
    float in_ = giS[row * 388 + 256 + c];
    float hr  = ghS[row * 388 + c];
    float hz  = ghS[row * 388 + 128 + c];
    float hn  = ghS[row * 388 + 256 + c];
    float r_ = 1.0f / (1.0f + expf(-(ir + hr)));
    float z_ = 1.0f / (1.0f + expf(-(iz + hz)));
    float n_ = tanhf(in_ + r_ * hn);
    float hold = hs[row * 132 + c];
    float hnew = (1.0f - z_) * n_ + z_ * hold;
    h[(size_t)(row0 + row) * 128 + c] = hnew;
    giS[row * 388 + c] = hnew;                 // stash for std pass
  }
  __syncthreads();

  // ---- per-row std (ddof=1), one wave per two rows ----
  for (int rr = wave; rr < 16; rr += 8) {
    float4 v = *(const float4*)&giS[rr * 388 + lane * 4];
    float s = v.x + v.y + v.z + v.w;
    float q = v.x * v.x + v.y * v.y + v.z * v.z + v.w * v.w;
    #pragma unroll
    for (int off = 16; off > 0; off >>= 1) {
      s += __shfl_xor(s, off, 32);
      q += __shfl_xor(q, off, 32);
    }
    float mean = s * (1.0f / 128.0f);
    float var  = (q - 128.0f * mean * mean) * (1.0f / 127.0f);
    float inv  = rsqrtf(var);
    float4 o; o.x = v.x * inv; o.y = v.y * inv; o.z = v.z * inv; o.w = v.w * inv;
    *(float4*)(xout + (size_t)(row0 + rr) * 128 + lane * 4) = o;
  }
}

// ---------------------------------------------------------------------------
// Input layout (setup_inputs dict order, params flattened by sorted keys):
//   0:x  1:edge_val  2:gru.b_hh  3:gru.b_ih  4:gru.w_hh  5:gru.w_ih
//   6..41: layers[L][b] -> W at 6+(L*3+b)*2, bias at +1   (L=0..5, b=0..2)
//   42:edge_row  43:edge_col  44:n_nodes
// ---------------------------------------------------------------------------
extern "C" void kernel_launch(void* const* d_in, const int* in_sizes, int n_in,
                              void* d_out, int out_size, void* d_ws, size_t ws_size,
                              hipStream_t stream) {
  (void)in_sizes; (void)n_in; (void)out_size; (void)ws_size;
  const float* x_in  = (const float*)d_in[0];
  const float* eval_ = (const float*)d_in[1];
  const float* b_hh  = (const float*)d_in[2];
  const float* b_ih  = (const float*)d_in[3];
  const float* w_hh  = (const float*)d_in[4];
  const float* w_ih  = (const float*)d_in[5];
  const int*   erow  = (const int*)d_in[42];
  const int*   ecol  = (const int*)d_in[43];

  auto Wof = [&](int L, int b) { return (const float*)d_in[6 + (L * 3 + b) * 2]; };
  auto Bof = [&](int L, int b) { return (const float*)d_in[6 + (L * 3 + b) * 2 + 1]; };

  // ---- pack all weights to f16 WMMA-B layout in workspace ----
  _Float16* Wpk = (_Float16*)d_ws;
  size_t woff = 0;
  size_t layerOff[6][3];
  for (int L = 0; L < 6; ++L)
    for (int b = 0; b < 3; ++b) {
      int Din = (b == 0 && (L % 3) == 2) ? 256 : 128;
      layerOff[L][b] = woff;
      int total = Din * 128;
      pack_w<<<(total + 255) / 256, 256, 0, stream>>>(Wof(L, b), Wpk + woff, Din / 32, 8, 0);
      woff += (size_t)total;
    }
  size_t ihOff = woff;   // w_ih^T : 128 x 384
  pack_w<<<(49152 + 255) / 256, 256, 0, stream>>>(w_ih, Wpk + ihOff, 4, 24, 1);
  woff += 49152;
  size_t hhOff = woff;   // w_hh^T : 128 x 384
  pack_w<<<(49152 + 255) / 256, 256, 0, stream>>>(w_hh, Wpk + hhOff, 4, 24, 1);
  woff += 49152;

  // ---- activation buffers in workspace ----
  float* act = (float*)((char*)d_ws + ((woff * 2 + 255) & ~(size_t)255));
  const size_t NF = (size_t)N_NODES * 128;
  float* hbuf = act;
  float* t0   = act + NF;
  float* agg  = act + 2 * NF;
  float* xbuf = act + 3 * NF;     // depth-0 normalized output

  const int nI = (int)NF;
  zero_f<<<(nI + 255) / 256, 256, 0, stream>>>(hbuf, nI);

  dim3 gTile(ROWTILES);
  const int scatterBlocks = (NEDGE * 32 + 255) / 256;

  for (int dep = 0; dep < 2; ++dep) {
    int L0 = 3 * dep, L1 = 3 * dep + 1, L2 = 3 * dep + 2;
    const float* xin = (dep == 0) ? x_in : xbuf;
    float*       xo  = (dep == 1) ? (float*)d_out : xbuf;

    // mlp0: x -> t0 (fused 3 layers)
    mlp3_fused<<<gTile, 256, 0, stream>>>(
        xin,
        Wpk + layerOff[L0][0], Bof(L0, 0),
        Wpk + layerOff[L0][1], Bof(L0, 1),
        Wpk + layerOff[L0][2], Bof(L0, 2),
        t0);

    // agg = A @ t0 (scatter-add over edges)
    zero_f<<<(nI + 255) / 256, 256, 0, stream>>>(agg, nI);
    scatter_edges<<<scatterBlocks, 256, 0, stream>>>(t0, erow, ecol, eval_, agg, NEDGE);

    // mlp1 + concat + mlp2 + GRU + std, fully fused
    mega_fused<<<gTile, 256, 0, stream>>>(
        xin, agg, hbuf,
        Wpk + layerOff[L1][0], Bof(L1, 0),
        Wpk + layerOff[L1][1], Bof(L1, 1),
        Wpk + layerOff[L1][2], Bof(L1, 2),
        Wpk + layerOff[L2][0], Bof(L2, 0),
        Wpk + layerOff[L2][1], Bof(L2, 1),
        Wpk + layerOff[L2][2], Bof(L2, 2),
        Wpk + ihOff, Wpk + hhOff, b_ih, b_hh,
        xo);
  }
}
